// LinearAttentionLayer_6837587935578
// MI455X (gfx1250) — compile-verified, matrix operand order checked by reading, not run on hardware
//
#include <hip/hip_runtime.h>
#include <math.h>

#define B_ 8
#define N_ 8192
#define L_ 8192
#define C_ 256
#define H_ 8
#define D_ 32

typedef __attribute__((ext_vector_type(16))) __bf16 v16bf;
typedef __attribute__((ext_vector_type(8)))  float  v8f;
typedef __attribute__((ext_vector_type(4)))  __bf16 bf16x4;
typedef __attribute__((ext_vector_type(2)))  __bf16 bf16x2;

// ---------------------------------------------------------------- utilities
__global__ __launch_bounds__(256) void cast_w(const float* __restrict__ in,
                                              __bf16* __restrict__ out, int n) {
  for (int i = blockIdx.x * 256 + threadIdx.x; i < n; i += gridDim.x * 256)
    out[i] = (__bf16)in[i];
}

// LayerNorm over C=256; one block (8 waves) per row; writes bf16.
__global__ __launch_bounds__(256) void ln_bf16(const float* __restrict__ x,
                                               const float* __restrict__ g,
                                               const float* __restrict__ b,
                                               __bf16* __restrict__ out) {
  int row = blockIdx.x, tid = threadIdx.x;
  float v = x[(size_t)row * C_ + tid];
  float s = v, s2 = v * v;
  for (int o = 16; o > 0; o >>= 1) {
    s  += __shfl_xor(s,  o, 32);
    s2 += __shfl_xor(s2, o, 32);
  }
  __shared__ float ps[8], ps2[8];
  int wave = tid >> 5, lane = tid & 31;
  if (lane == 0) { ps[wave] = s; ps2[wave] = s2; }
  __syncthreads();
  if (tid == 0) {
    float a = 0.f, a2 = 0.f;
    for (int i = 0; i < 8; ++i) { a += ps[i]; a2 += ps2[i]; }
    ps[0] = a; ps2[0] = a2;
  }
  __syncthreads();
  float mu  = ps[0] * (1.f / C_);
  float var = ps2[0] * (1.f / C_) - mu * mu;
  float r   = rsqrtf(var + 1e-5f);
  out[(size_t)row * C_ + tid] = (__bf16)((v - mu) * r * g[tid] + b[tid]);
}

// ---------------------------------------------------------------- WMMA GEMM
// C = A(MxK,bf16) * B(KxN,bf16), f32 accumulate, fused epilogues.
// Block tile 128x128, 8 waves (4x2), each wave 32x64 = 2x4 WMMA 16x16 tiles.
// Double-buffered LDS, register-staged global loads, 1 barrier / K-step.
enum { EPI_Q = 0, EPI_KV = 1, EPI_BIAS_RES = 2, EPI_BIAS_GELU = 3 };

__device__ __forceinline__ void load_tiles(const __bf16* __restrict__ A,
                                           const __bf16* __restrict__ Bw,
                                           int N, int K, int bm0, int bn0,
                                           int kt, int tid,
                                           bf16x4 ra[4], bf16x4 rb[4]) {
#pragma unroll
  for (int i = 0; i < 4; ++i) {        // A tile 128x32
    int c = tid + 256 * i;
    int r = c >> 3, c4 = (c & 7) << 2;
    ra[i] = *(const bf16x4*)&A[(size_t)(bm0 + r) * K + kt + c4];
  }
#pragma unroll
  for (int i = 0; i < 4; ++i) {        // B tile 32x128
    int c = tid + 256 * i;
    int r = c >> 5, c4 = (c & 31) << 2;
    rb[i] = *(const bf16x4*)&Bw[(size_t)(kt + r) * N + bn0 + c4];
  }
}

__device__ __forceinline__ void store_tiles(__bf16 (*sA)[40], __bf16 (*sBt)[40],
                                            int tid, const bf16x4 ra[4],
                                            const bf16x4 rb[4]) {
#pragma unroll
  for (int i = 0; i < 4; ++i) {
    int c = tid + 256 * i;
    int r = c >> 3, c4 = (c & 7) << 2;
    *(bf16x4*)&sA[r][c4] = ra[i];
  }
#pragma unroll
  for (int i = 0; i < 4; ++i) {        // transpose into sBt[n][k]
    int c = tid + 256 * i;
    int r = c >> 5, c4 = (c & 31) << 2;
    sBt[c4 + 0][r] = rb[i][0]; sBt[c4 + 1][r] = rb[i][1];
    sBt[c4 + 2][r] = rb[i][2]; sBt[c4 + 3][r] = rb[i][3];
  }
}

template <int EPI>
__global__ __launch_bounds__(256) void gemm_bf16_epi(
    const __bf16* __restrict__ A, const __bf16* __restrict__ Bw,
    int M, int N, int K,
    const float* __restrict__ bias, const float* __restrict__ rowscale,
    const float* __restrict__ resid, float vscale,
    float* __restrict__ outF, __bf16* __restrict__ outB,
    __bf16* __restrict__ outB2) {
  // pads: +8 bf16 (16B) keep 8B vector alignment and stagger banks
  __shared__ __bf16 sA[2][128][40];
  __shared__ __bf16 sBt[2][128][40];   // B tile stored transposed: [n][k]

  const int tid  = threadIdx.x;
  const int wave = tid >> 5, lane = tid & 31;
  const int wm = wave & 3, wn = wave >> 2;   // wave grid 4(M) x 2(N)
  const int lm = lane & 15, lk = lane >> 4;  // fragment lane decomposition
  const int bm0 = blockIdx.y * 128, bn0 = blockIdx.x * 128;

  v8f acc[2][4] = {};
  bf16x4 ra[4], rb[4];

  // prologue: stage K-tile 0
  load_tiles(A, Bw, N, K, bm0, bn0, 0, tid, ra, rb);
  store_tiles(sA[0], sBt[0], tid, ra, rb);
  __syncthreads();

  int pb = 0;
  for (int kt = 0; kt < K; kt += 32) {
    const bool more = (kt + 32) < K;
    if (more) {                       // issue next tile's global loads
      load_tiles(A, Bw, N, K, bm0, bn0, kt + 32, tid, ra, rb);
      if (kt + 64 < K) {              // pull tile after next toward the WGP
        __builtin_prefetch(&A[(size_t)(bm0 + (tid >> 1)) * K + kt + 64], 0, 1);
        __builtin_prefetch(&Bw[(size_t)(kt + 64 + (tid >> 3)) * N + bn0], 0, 1);
      }
    }

    // Fragment gather per documented 16-bit layouts:
    // lanes 0-15 hold K={0..7,16..23}, lanes 16-31 hold K={8..15,24..31}
    v16bf af[2], bfr[4];
#pragma unroll
    for (int mt = 0; mt < 2; ++mt) {
      int row = wm * 32 + mt * 16 + lm;
#pragma unroll
      for (int v = 0; v < 8; ++v) {
        int kk = ((v < 4) ? (v << 1) : (16 + ((v - 4) << 1))) + (lk << 3);
        bf16x2 p = *(const bf16x2*)&sA[pb][row][kk];
        af[mt][2 * v] = p[0]; af[mt][2 * v + 1] = p[1];
      }
    }
#pragma unroll
    for (int nt = 0; nt < 4; ++nt) {
      int coln = wn * 64 + nt * 16 + lm;
#pragma unroll
      for (int v = 0; v < 8; ++v) {
        int kk = ((v < 4) ? (v << 1) : (16 + ((v - 4) << 1))) + (lk << 3);
        bf16x2 p = *(const bf16x2*)&sBt[pb][coln][kk];
        bfr[nt][2 * v] = p[0]; bfr[nt][2 * v + 1] = p[1];
      }
    }
#pragma unroll
    for (int mt = 0; mt < 2; ++mt)
#pragma unroll
      for (int nt = 0; nt < 4; ++nt)
        acc[mt][nt] = __builtin_amdgcn_wmma_f32_16x16x32_bf16(
            false, af[mt], false, bfr[nt], (short)0, acc[mt][nt], false, false);

    if (more) store_tiles(sA[pb ^ 1], sBt[pb ^ 1], tid, ra, rb);
    __syncthreads();
    pb ^= 1;
  }

  // D layout: VGPR v -> M = 8*(lane/16)+v, N = lane%16
#pragma unroll
  for (int mt = 0; mt < 2; ++mt)
#pragma unroll
    for (int nt = 0; nt < 4; ++nt)
#pragma unroll
      for (int v = 0; v < 8; ++v) {
        int row = bm0 + wm * 32 + mt * 16 + (lk << 3) + v;
        int col = bn0 + wn * 64 + nt * 16 + lm;
        float a = acc[mt][nt][v];
        if (EPI == EPI_Q) {                       // (elu+1) * q_mask -> bf16
          float e = (a > 0.f) ? (a + 1.f) : __expf(a);
          outB[(size_t)row * N + col] = (__bf16)(e * rowscale[row]);
        } else if (EPI == EPI_KV) {               // split k|v halves
          if (col < 256) {
            float e = (a > 0.f) ? (a + 1.f) : __expf(a);
            outB[(size_t)row * 256 + col] = (__bf16)(e * rowscale[row]);
          } else {
            outB2[(size_t)row * 256 + (col - 256)] =
                (__bf16)(a * rowscale[row] * vscale);
          }
        } else if (EPI == EPI_BIAS_RES) {         // + bias + residual -> f32
          outF[(size_t)row * N + col] =
              a + bias[col] + resid[(size_t)row * N + col];
        } else {                                  // + bias, exact GELU -> bf16
          float t  = a + bias[col];
          float ge = 0.5f * t * (1.f + erff(t * 0.70710678118f));
          outB[(size_t)row * N + col] = (__bf16)ge;
        }
      }
}

// ------------------------------------------------- kv_state + k_sum (BW-bound)
// One block per (b,h): kv_state[m][d] = sum_l v[l,m]*k[l,d]; ksum[d] = sum_l k
__global__ __launch_bounds__(256) void kvstate_ker(const __bf16* __restrict__ kbuf,
                                                   const __bf16* __restrict__ vbuf,
                                                   float* __restrict__ kvs,
                                                   float* __restrict__ ksum) {
  int bh = blockIdx.x, b = bh / H_, h = bh % H_;
  int tid = threadIdx.x;
  __shared__ __bf16 sk[64][33];
  __shared__ __bf16 sv[64][33];
  float acc[4] = {0.f, 0.f, 0.f, 0.f};
  float ks = 0.f;
  int mi[4], di[4];
#pragma unroll
  for (int i = 0; i < 4; ++i) { int o = i * 256 + tid; mi[i] = o >> 5; di[i] = o & 31; }
  for (int l0 = 0; l0 < L_; l0 += 64) {
#pragma unroll
    for (int i = 0; i < 8; ++i) {
      int e = tid + 256 * i;
      int r = e >> 5, c = e & 31;
      size_t g = ((size_t)(b * L_ + l0 + r)) * C_ + h * D_ + c;
      sk[r][c] = kbuf[g];
      sv[r][c] = vbuf[g];
    }
    __syncthreads();
    for (int l = 0; l < 64; ++l) {
#pragma unroll
      for (int i = 0; i < 4; ++i)
        acc[i] += (float)sk[l][di[i]] * (float)sv[l][mi[i]];
      if (tid < 32) ks += (float)sk[l][tid];
    }
    __syncthreads();
  }
#pragma unroll
  for (int i = 0; i < 4; ++i)
    kvs[((size_t)bh * D_ + mi[i]) * D_ + di[i]] = acc[i];
  if (tid < 32) ksum[(size_t)bh * D_ + tid] = ks;
}

// ------------------------------------------------- z = 1/(q . ksum + 1e-6)
__global__ __launch_bounds__(256) void z_ker(const __bf16* __restrict__ qv,
                                             const float* __restrict__ ksum,
                                             float* __restrict__ z) {
  int idx = blockIdx.x * 256 + threadIdx.x;   // over B*N*H
  int row = idx >> 3, h = idx & 7;
  int b = row / N_;
  const __bf16* qp = qv + (size_t)row * C_ + h * D_;
  const float*  kp = ksum + ((size_t)(b * H_ + h)) * D_;
  float d = 0.f;
#pragma unroll
  for (int i = 0; i < 32; ++i) d += (float)qp[i] * kp[i];
  z[idx] = 1.f / (d + 1e-6f);
}

// -------------------------- attn[n,h,m] = L * z * sum_d q[d]*kv_state[h,m,d]
__global__ __launch_bounds__(256) void apply_ker(const __bf16* __restrict__ qv,
                                                 const float* __restrict__ kvs,
                                                 const float* __restrict__ z,
                                                 __bf16* __restrict__ attn) {
  int r0 = blockIdx.x * 8;
  int b  = r0 / N_;
  int tid = threadIdx.x;
  __shared__ float  skv[H_ * D_ * D_];   // 32 KB: all heads of this batch
  __shared__ __bf16 sq[8][C_];
  __shared__ float  sz[64];
#pragma unroll
  for (int i = 0; i < 32; ++i)
    skv[tid + 256 * i] = kvs[(size_t)b * H_ * D_ * D_ + tid + 256 * i];
#pragma unroll
  for (int i = 0; i < 8; ++i)
    sq[i][tid] = qv[(size_t)(r0 + i) * C_ + tid];
  if (tid < 64) sz[tid] = z[(size_t)(r0 + (tid >> 3)) * H_ + (tid & 7)];
  __syncthreads();
  int h = tid >> 5, m = tid & 31;        // col = tid = h*32+m
  const float* kh = &skv[h * D_ * D_ + m * D_];
#pragma unroll
  for (int r = 0; r < 8; ++r) {
    float a = 0.f;
#pragma unroll
    for (int d = 0; d < 32; ++d) a += (float)sq[r][h * D_ + d] * kh[d];
    attn[(size_t)(r0 + r) * C_ + tid] = (__bf16)(a * sz[r * 8 + h] * (float)L_);
  }
}

// ---------------------------------------------------------------- launcher
extern "C" void kernel_launch(void* const* d_in, const int* in_sizes, int n_in,
                              void* d_out, int out_size, void* d_ws, size_t ws_size,
                              hipStream_t stream) {
  const float* q      = (const float*)d_in[0];
  const float* kv     = (const float*)d_in[1];
  const float* qmask  = (const float*)d_in[2];
  const float* kvmask = (const float*)d_in[3];
  const float* Wq     = (const float*)d_in[4];
  const float* Wkv    = (const float*)d_in[5];
  const float* Wp     = (const float*)d_in[6];
  const float* bp     = (const float*)d_in[7];
  const float* ln1g   = (const float*)d_in[8];
  const float* ln1b   = (const float*)d_in[9];
  const float* ln2g   = (const float*)d_in[10];
  const float* ln2b   = (const float*)d_in[11];
  const float* W1     = (const float*)d_in[12];
  const float* b1     = (const float*)d_in[13];
  const float* W2     = (const float*)d_in[14];
  const float* b2     = (const float*)d_in[15];
  float* out = (float*)d_out;
  (void)in_sizes; (void)n_in; (void)out_size; (void)ws_size;

  char* ws = (char*)d_ws;
  size_t off = 0;
  auto alloc = [&](size_t bytes) {
    void* p = ws + off;
    off += (bytes + 255) & ~(size_t)255;
    return p;
  };
  const size_t R = (size_t)B_ * N_;                     // 65536 rows
  __bf16* bufA = (__bf16*)alloc(R * C_ * 2);            // qn -> attn (reused)
  __bf16* bufB = (__bf16*)alloc(R * C_ * 2);            // kvn -> ln2x (reused)
  __bf16* qvb  = (__bf16*)alloc(R * C_ * 2);            // q_vec
  __bf16* kb   = (__bf16*)alloc(R * C_ * 2);            // k_vec -> h1 lo (reused)
  __bf16* vb   = (__bf16*)alloc(R * C_ * 2);            // v_vec -> h1 hi (contig)
  float*  xf   = (float*)alloc(R * C_ * 4);             // residual x (f32)
  __bf16* Wqb  = (__bf16*)alloc((size_t)C_ * C_ * 2);
  __bf16* Wkvb = (__bf16*)alloc((size_t)C_ * 2 * C_ * 2);
  __bf16* Wpb  = (__bf16*)alloc((size_t)C_ * C_ * 2);
  __bf16* W1b  = (__bf16*)alloc((size_t)C_ * 2 * C_ * 2);
  __bf16* W2b  = (__bf16*)alloc((size_t)2 * C_ * C_ * 2);
  float*  kvs  = (float*)alloc((size_t)B_ * H_ * D_ * D_ * 4);
  float*  ksum = (float*)alloc((size_t)B_ * H_ * D_ * 4);
  float*  zbuf = (float*)alloc(R * H_ * 4);
  __bf16* h1   = kb;   // (B,N,2C) bf16 spans kb..vb (allocated contiguously)

  // weights -> bf16
  cast_w<<<128, 256, 0, stream>>>(Wq,  Wqb,  C_ * C_);
  cast_w<<<256, 256, 0, stream>>>(Wkv, Wkvb, C_ * 2 * C_);
  cast_w<<<128, 256, 0, stream>>>(Wp,  Wpb,  C_ * C_);
  cast_w<<<256, 256, 0, stream>>>(W1,  W1b,  C_ * 2 * C_);
  cast_w<<<256, 256, 0, stream>>>(W2,  W2b,  2 * C_ * C_);

  // LN1 on q and kv
  ln_bf16<<<(int)R, 256, 0, stream>>>(q,  ln1g, ln1b, bufA);
  ln_bf16<<<(int)R, 256, 0, stream>>>(kv, ln1g, ln1b, bufB);

  // q_vec = (elu(qn@Wq)+1) * q_mask
  gemm_bf16_epi<EPI_Q><<<dim3(C_ / 128, (int)R / 128), 256, 0, stream>>>(
      bufA, Wqb, (int)R, C_, C_, nullptr, qmask, nullptr, 0.f,
      nullptr, qvb, nullptr);

  // k_vec/v_vec from kvn@Wkv (v pre-scaled by 1/L)
  gemm_bf16_epi<EPI_KV><<<dim3(2 * C_ / 128, (int)R / 128), 256, 0, stream>>>(
      bufB, Wkvb, (int)R, 2 * C_, C_, nullptr, kvmask, nullptr,
      1.f / (float)L_, nullptr, kb, vb);

  // kv_state + k_sum per (b,h)
  kvstate_ker<<<B_ * H_, 256, 0, stream>>>(kb, vb, kvs, ksum);

  // z normalizer
  z_ker<<<(int)(R * H_ / 256), 256, 0, stream>>>(qvb, ksum, zbuf);

  // apply attention state
  apply_ker<<<(int)(R / 8), 256, 0, stream>>>(qvb, kvs, zbuf, bufA);

  // x = q + attn@Wp + bp
  gemm_bf16_epi<EPI_BIAS_RES><<<dim3(C_ / 128, (int)R / 128), 256, 0, stream>>>(
      bufA, Wpb, (int)R, C_, C_, bp, nullptr, q, 0.f, xf, nullptr, nullptr);

  // LN2
  ln_bf16<<<(int)R, 256, 0, stream>>>(xf, ln2g, ln2b, bufB);

  // h1 = gelu(ln2x@W1 + b1)
  gemm_bf16_epi<EPI_BIAS_GELU><<<dim3(2 * C_ / 128, (int)R / 128), 256, 0, stream>>>(
      bufB, W1b, (int)R, 2 * C_, C_, b1, nullptr, nullptr, 0.f,
      nullptr, h1, nullptr);

  // out = x + h1@W2 + b2
  gemm_bf16_epi<EPI_BIAS_RES><<<dim3(C_ / 128, (int)R / 128), 256, 0, stream>>>(
      h1, W2b, (int)R, C_, 2 * C_, b2, nullptr, xf, 0.f, out, nullptr, nullptr);
}
